// MambaMineSimple_25108378812676
// MI455X (gfx1250) — matrix operand order repaired and checked by reading, not compiled, not run on hardware
//
#include <hip/hip_runtime.h>

// ---------------------------------------------------------------------------
// Mamba2 block forward for gfx1250 (MI455X).
// All GEMM-like stages use v_wmma_f32_16x16x32_bf16 (wave32, fp32 accum).
// ---------------------------------------------------------------------------

typedef __bf16 bf16;
typedef __attribute__((ext_vector_type(8)))  bf16  bf16x8;
typedef __attribute__((ext_vector_type(16))) bf16  bf16x16;
typedef __attribute__((ext_vector_type(8)))  float floatx8;

#define TOKENS   8192      // B*L
#define DMODEL   1024
#define DINNER   2048
#define DSTATE   128
#define NHEAD    32
#define PHEAD    64
#define DPROJ    4384      // 2*2048 + 2*128 + 32
#define CONVD    2304      // 2048 + 256
#define NCHUNK   32        // B * (L/256)
#define CHUNK    256

// ---- WMMA helpers (layouts per CDNA5 ISA 7.12.2, wave32) -------------------

// A operand 16x32 bf16: lane holds row m=lane%16; half=lane/16.
// v16 elems 0..7  <-> K = half*8 + e        (contiguous 8)
// v16 elems 8..15 <-> K = 16 + half*8 + e-8 (contiguous 8, offset +16)
__device__ __forceinline__ bf16x16 frag_a_load(const bf16* base, int ld) {
    int lane = threadIdx.x & 31;
    int row = lane & 15, half = lane >> 4;
    const bf16* p = base + (long)row * ld + half * 8;
    bf16x8 lo = *(const bf16x8*)(p);
    bf16x8 hi = *(const bf16x8*)(p + 16);
    bf16x16 f;
#pragma unroll
    for (int i = 0; i < 8; ++i) { f[i] = lo[i]; f[i + 8] = hi[i]; }
    return f;
}

// B operand 32x16 bf16: lane holds col n=lane%16; half=lane/16.
// v16 elems 0..7 <-> K = half*8 + e; elems 8..15 <-> K = 16 + half*8 + e-8. Memory must be
// laid out "K-contiguous per output column" (i.e. transposed / [n][k]).
__device__ __forceinline__ bf16x16 frag_b_load(const bf16* base, int ld) {
    int lane = threadIdx.x & 31;
    int col = lane & 15, half = lane >> 4;
    const bf16* p = base + (long)col * ld + half * 8;
    bf16x8 lo = *(const bf16x8*)(p);
    bf16x8 hi = *(const bf16x8*)(p + 16);
    bf16x16 f;
#pragma unroll
    for (int i = 0; i < 8; ++i) { f[i] = lo[i]; f[i + 8] = hi[i]; }
    return f;
}

// Same, from fp32 memory with on-the-fly cvt.
__device__ __forceinline__ bf16x16 frag_b_load_f32(const float* base, int ld) {
    int lane = threadIdx.x & 31;
    int col = lane & 15, half = lane >> 4;
    const float* p = base + (long)col * ld + half * 8;
    bf16x16 f;
#pragma unroll
    for (int i = 0; i < 16; ++i) f[i] = (bf16)p[i < 8 ? i : i + 8];
    return f;
}

__device__ __forceinline__ floatx8 wmma_bf16(bf16x16 a, bf16x16 b, floatx8 c) {
    return __builtin_amdgcn_wmma_f32_16x16x32_bf16(false, a, false, b,
                                                   (short)0, c, false, false);
}

// C/D 16x16 f32: lane col n=lane%16, vgpr r -> row = (lane/16)*8 + r.
__device__ __forceinline__ void frag_store(float* base, int ld, floatx8 acc) {
    int lane = threadIdx.x & 31;
    int col = lane & 15, half = lane >> 4;
#pragma unroll
    for (int r = 0; r < 8; ++r) base[(long)(half * 8 + r) * ld + col] = acc[r];
}

// ---- elementwise fp32 -> bf16 ---------------------------------------------
__global__ void cvt_bf16_k(const float* __restrict__ src, bf16* __restrict__ dst, long n) {
    long i = (long)blockIdx.x * 256 + threadIdx.x;
    if (i < n) dst[i] = (bf16)src[i];
}

// ---- generic GEMM: C[M,N] = A[M,K](bf16) * W[N,K]^T(bf16), fp32 out -------
// block = 256 threads = 8 waves; wave computes 64(M) x 32(N); grid (M/512, N/32)
// Per K-step: 8 WMMA vs 12 b128 loads (~21 FLOP/byte) + prefetch of next tiles.
__global__ void __launch_bounds__(256)
gemm_bf16_k(const bf16* __restrict__ A, const bf16* __restrict__ W,
            float* __restrict__ C, int M, int N, int K) {
    int wave = threadIdx.x >> 5;
    int n0 = blockIdx.y * 32;
    int m0 = (blockIdx.x * 8 + wave) * 64;
    const bf16* wrow0 = W + (long)n0 * K;
    const bf16* wrow1 = W + (long)(n0 + 16) * K;
    const bf16* arow  = A + (long)m0 * K;
    floatx8 c00 = {}, c10 = {}, c20 = {}, c30 = {};
    floatx8 c01 = {}, c11 = {}, c21 = {}, c31 = {};
    for (int k = 0; k < K; k += 32) {
        // prefetch next K tiles into cache hierarchy (global_prefetch_b8)
        if (k + 32 < K) {
            __builtin_prefetch(arow  + k + 32, 0, 3);
            __builtin_prefetch(wrow0 + k + 32, 0, 3);
        }
        bf16x16 b0 = frag_b_load(wrow0 + k, K);
        bf16x16 b1 = frag_b_load(wrow1 + k, K);
        bf16x16 f0 = frag_a_load(arow + k, K);
        bf16x16 f1 = frag_a_load(arow + (long)16 * K + k, K);
        bf16x16 f2 = frag_a_load(arow + (long)32 * K + k, K);
        bf16x16 f3 = frag_a_load(arow + (long)48 * K + k, K);
        c00 = wmma_bf16(f0, b0, c00);
        c10 = wmma_bf16(f1, b0, c10);
        c20 = wmma_bf16(f2, b0, c20);
        c30 = wmma_bf16(f3, b0, c30);
        c01 = wmma_bf16(f0, b1, c01);
        c11 = wmma_bf16(f1, b1, c11);
        c21 = wmma_bf16(f2, b1, c21);
        c31 = wmma_bf16(f3, b1, c31);
    }
    frag_store(C + (long)(m0 +  0) * N + n0,      N, c00);
    frag_store(C + (long)(m0 + 16) * N + n0,      N, c10);
    frag_store(C + (long)(m0 + 32) * N + n0,      N, c20);
    frag_store(C + (long)(m0 + 48) * N + n0,      N, c30);
    frag_store(C + (long)(m0 +  0) * N + n0 + 16, N, c01);
    frag_store(C + (long)(m0 + 16) * N + n0 + 16, N, c11);
    frag_store(C + (long)(m0 + 32) * N + n0 + 16, N, c21);
    frag_store(C + (long)(m0 + 48) * N + n0 + 16, N, c31);
}

// ---- depthwise conv4 + SiLU + route into WMMA-friendly bf16 layouts -------
__global__ void conv_silu_k(const float* __restrict__ zx, const float* __restrict__ conv_w,
                            const float* __restrict__ conv_b,
                            bf16* __restrict__ xT, bf16* __restrict__ Bbf,
                            bf16* __restrict__ BT, bf16* __restrict__ Cbf) {
    long idx = (long)blockIdx.x * 256 + threadIdx.x;
    if (idx >= (long)TOKENS * CONVD) return;
    int ch = (int)(idx % CONVD);
    long tok = idx / CONVD;
    int l = (int)(tok & 4095), b = (int)(tok >> 12);
    float acc = conv_b[ch];
#pragma unroll
    for (int kk = 0; kk < 4; ++kk) {
        int ll = l - 3 + kk;
        if (ll >= 0)
            acc += conv_w[ch * 4 + kk] * zx[((long)(b * 4096 + ll)) * DPROJ + DINNER + ch];
    }
    float s = acc / (1.f + __expf(-acc));          // SiLU
    int c = l >> 8, j = l & 255, bc = b * 16 + c;
    if (ch < DINNER) {
        int h = ch >> 6, p = ch & 63;              // xT[bc][h][p][j]
        xT[(((long)(bc * NHEAD + h)) * PHEAD + p) * CHUNK + j] = (bf16)s;
    } else if (ch < DINNER + DSTATE) {
        int n = ch - DINNER;
        Bbf[tok * DSTATE + n] = (bf16)s;           // B row-major [tok][n]
        BT[((long)bc * DSTATE + n) * CHUNK + j] = (bf16)s;  // BT[bc][n][j]
    } else {
        int n = ch - DINNER - DSTATE;
        Cbf[tok * DSTATE + n] = (bf16)s;           // C row-major [tok][n]
    }
}

// ---- dt = softplus(dt_raw + dt_bias) --------------------------------------
__global__ void dt_k(const float* __restrict__ zx, const float* __restrict__ dt_bias,
                     float* __restrict__ dtv) {
    long i = (long)blockIdx.x * 256 + threadIdx.x;
    if (i >= (long)TOKENS * NHEAD) return;
    int h = (int)(i & 31);
    long tok = i >> 5;
    float x = zx[tok * DPROJ + (DINNER + CONVD) + h] + dt_bias[h];
    dtv[i] = (x > 20.f) ? x : log1pf(__expf(x));
}

// ---- per-(chunk,head) cumsum of a = dt*A, chunk decay, state weights ------
// Also emits dtc[bch][j] (contiguous dt per chunk-head) so y1_k's A-fragment
// build reads only contiguous streams.
__global__ void cumsum_k(const float* __restrict__ dtv, const float* __restrict__ A_log,
                         float* __restrict__ Acs, float* __restrict__ wdec,
                         float* __restrict__ cdec, float* __restrict__ dtc) {
    int bch = blockIdx.x;                // bc*32 + h
    int h = bch & 31, bc = bch >> 5;
    int t = threadIdx.x;
    long tok = (long)bc * CHUNK + t;
    float dtval = dtv[tok * NHEAD + h];
    dtc[(long)bch * CHUNK + t] = dtval;
    float a = dtval * (-__expf(A_log[h]));
    __shared__ float buf[2][CHUNK];
    int src = 0;
    buf[0][t] = a; __syncthreads();
    for (int off = 1; off < CHUNK; off <<= 1) {
        float v = buf[src][t] + ((t >= off) ? buf[src][t - off] : 0.f);
        buf[1 - src][t] = v; src = 1 - src; __syncthreads();
    }
    float cs = buf[src][t];
    Acs[(long)bch * CHUNK + t] = cs;
    __shared__ float lastv;
    if (t == CHUNK - 1) { lastv = cs; cdec[bch] = __expf(cs); }
    __syncthreads();
    wdec[(long)bch * CHUNK + t] = dtval * __expf(lastv - cs);
}

// ---- CB[bc] = Cc (256x128) * Bc^T (128x256), shared across heads ----------
__global__ void cb_k(const bf16* __restrict__ Cbf, const bf16* __restrict__ Bbf,
                     float* __restrict__ CB) {
    int bc = blockIdx.x, wave = threadIdx.x >> 5;
    int tile = blockIdx.y * 8 + wave;    // 256 tiles of 16x16
    int it = tile >> 4, jt = tile & 15;
    const bf16* Ab = Cbf + ((long)bc * CHUNK + it * 16) * DSTATE;
    const bf16* Bb = Bbf + ((long)bc * CHUNK + jt * 16) * DSTATE;
    floatx8 acc = {};
    for (int k = 0; k < DSTATE; k += 32)
        acc = wmma_bf16(frag_a_load(Ab + k, DSTATE), frag_b_load(Bb + k, DSTATE), acc);
    frag_store(CB + ((long)bc * CHUNK + it * 16) * CHUNK + jt * 16, CHUNK, acc);
}

// ---- intra-chunk y1 = scores @ x (scores built on the fly, never stored) --
__global__ void y1_k(const float* __restrict__ CB, const float* __restrict__ Acs,
                     const float* __restrict__ dtc, const bf16* __restrict__ xT,
                     float* __restrict__ Y) {
    int bc = blockIdx.x, h = blockIdx.y;
    int wave = threadIdx.x >> 5;
    int tile = blockIdx.z * 8 + wave;    // 64 tiles: 16 i-tiles x 4 p-tiles
    int it = tile >> 2, pt = tile & 3;
    int lane = threadIdx.x & 31, rowl = lane & 15, half = lane >> 4;
    int gi = it * 16 + rowl;
    int bch = bc * NHEAD + h;
    float acsI = Acs[(long)bch * CHUNK + gi];
    const float* acsJ  = Acs + (long)bch * CHUNK;
    const float* dtJ   = dtc + (long)bch * CHUNK;
    const float* cbrow = CB + ((long)bc * CHUNK + gi) * CHUNK;
    const bf16* xbase = xT + (((long)bch) * PHEAD + pt * 16) * CHUNK;
    floatx8 acc = {};
    for (int k0 = 0; k0 < CHUNK; k0 += 32) {
        if (k0 > it * 16 + 15) break;    // wave-uniform causal skip (EXEC stays full)
        bf16x16 a;
#pragma unroll
        for (int e = 0; e < 16; ++e) {
            int j = k0 + (e < 8 ? half * 8 + e : 16 + half * 8 + (e - 8));
            float v = 0.f;
            if (j <= gi)
                v = cbrow[j] * __expf(acsI - acsJ[j]) * dtJ[j];
            a[e] = (bf16)v;
        }
        acc = wmma_bf16(a, frag_b_load(xbase + k0, CHUNK), acc);
    }
#pragma unroll
    for (int r = 0; r < 8; ++r)
        Y[((long)bc * CHUNK + it * 16 + half * 8 + r) * DINNER + h * PHEAD + pt * 16 + rowl] = acc[r];
}

// ---- per-chunk states[h] = (wdec * x)(64x256) @ B(256x128) ----------------
__global__ void states_k(const bf16* __restrict__ xT, const bf16* __restrict__ BT,
                         const float* __restrict__ wdec, float* __restrict__ S) {
    int bc = blockIdx.x, h = blockIdx.y, bch = bc * NHEAD + h;
    int wave = threadIdx.x >> 5;
    int lane = threadIdx.x & 31, rowl = lane & 15, half = lane >> 4;
    const float* w = wdec + (long)bch * CHUNK;
    for (int tt = 0; tt < 4; ++tt) {
        int tile = tt * 8 + wave;        // 32 tiles: 4 p-tiles x 8 n-tiles
        int mt = tile >> 3, nt = tile & 7;
        int p = mt * 16 + rowl;
        const bf16* xrow = xT + (((long)bch) * PHEAD + p) * CHUNK;
        const bf16* bbase = BT + ((long)bc * DSTATE + nt * 16) * CHUNK;
        floatx8 acc = {};
        for (int k0 = 0; k0 < CHUNK; k0 += 32) {
            bf16x16 a;
#pragma unroll
            for (int e = 0; e < 16; ++e) {
                int j = k0 + (e < 8 ? half * 8 + e : 16 + half * 8 + (e - 8));
                a[e] = (bf16)((float)xrow[j] * w[j]);
            }
            acc = wmma_bf16(a, frag_b_load(bbase + k0, CHUNK), acc);
        }
        frag_store(S + (((long)bch) * PHEAD + mt * 16) * DSTATE + nt * 16, DSTATE, acc);
    }
}

// ---- inter-chunk sequential scan (16 steps) -------------------------------
__global__ void scan_k(const float* __restrict__ S, const float* __restrict__ cdec,
                       float* __restrict__ P) {
    long idx = (long)blockIdx.x * 256 + threadIdx.x;      // b,h,p,n
    if (idx >= (long)2 * NHEAD * PHEAD * DSTATE) return;
    int n = (int)(idx & 127), p = (int)((idx >> 7) & 63);
    int h = (int)((idx >> 13) & 31), b = (int)(idx >> 18);
    float carry = 0.f;
    for (int c = 0; c < 16; ++c) {
        int bch = (b * 16 + c) * NHEAD + h;
        long off = (((long)bch) * PHEAD + p) * DSTATE + n;
        P[off] = carry;                                   // state entering chunk c
        carry = carry * cdec[bch] + S[off];
    }
}

// ---- y += exp(Acs_i) * ( C(256x128) @ prevs[h]^T(128x64) ) ----------------
__global__ void y2_k(const bf16* __restrict__ Cbf, const float* __restrict__ P,
                     const float* __restrict__ Acs, float* __restrict__ Y) {
    int bc = blockIdx.x, h = blockIdx.y, bch = bc * NHEAD + h;
    int wave = threadIdx.x >> 5;
    int tile = blockIdx.z * 8 + wave;
    int it = tile >> 2, pt = tile & 3;
    int lane = threadIdx.x & 31, rowl = lane & 15, half = lane >> 4;
    const bf16* arow = Cbf + ((long)bc * CHUNK + it * 16) * DSTATE;
    const float* bbase = P + (((long)bch) * PHEAD + pt * 16) * DSTATE;
    floatx8 acc = {};
    for (int k0 = 0; k0 < DSTATE; k0 += 32)
        acc = wmma_bf16(frag_a_load(arow + k0, DSTATE), frag_b_load_f32(bbase + k0, DSTATE), acc);
#pragma unroll
    for (int r = 0; r < 8; ++r) {
        int gi = it * 16 + half * 8 + r;
        float f = __expf(Acs[(long)bch * CHUNK + gi]);
        Y[((long)bc * CHUNK + gi) * DINNER + h * PHEAD + pt * 16 + rowl] += f * acc[r];
    }
}

// ---- + D*x, gate by silu(z), RMSNorm, emit bf16 for out_proj --------------
__global__ void final_k(const float* __restrict__ Y, const bf16* __restrict__ xT,
                        const float* __restrict__ zx, const float* __restrict__ Dv,
                        const float* __restrict__ normw, bf16* __restrict__ out) {
    int tok = blockIdx.x, t = threadIdx.x;
    int b = tok >> 12, l = tok & 4095;
    int bc = b * 16 + (l >> 8), j = l & 255;
    float vals[8]; float ss = 0.f;
#pragma unroll
    for (int q = 0; q < 8; ++q) {
        int ch = q * 256 + t;
        int h = ch >> 6, p = ch & 63;
        float xval = (float)xT[(((long)(bc * NHEAD + h)) * PHEAD + p) * CHUNK + j];
        float y = Y[(long)tok * DINNER + ch] + Dv[h] * xval;
        float z = zx[(long)tok * DPROJ + ch];
        y *= z / (1.f + __expf(-z));
        vals[q] = y; ss += y * y;
    }
    __shared__ float red[256];
    red[t] = ss; __syncthreads();
    for (int off = 128; off > 0; off >>= 1) {
        if (t < off) red[t] += red[t + off];
        __syncthreads();
    }
    float rms = rsqrtf(red[0] / (float)DINNER + 1e-5f);
#pragma unroll
    for (int q = 0; q < 8; ++q) {
        int ch = q * 256 + t;
        out[(long)tok * DINNER + ch] = (bf16)(vals[q] * rms * normw[ch]);
    }
}

// ---------------------------------------------------------------------------
extern "C" void kernel_launch(void* const* d_in, const int* in_sizes, int n_in,
                              void* d_out, int out_size, void* d_ws, size_t ws_size,
                              hipStream_t stream) {
    const float* u          = (const float*)d_in[0];
    const float* in_proj_w  = (const float*)d_in[1];
    const float* conv_w     = (const float*)d_in[2];
    const float* conv_b     = (const float*)d_in[3];
    const float* dt_bias    = (const float*)d_in[4];
    const float* A_log      = (const float*)d_in[5];
    const float* Dv         = (const float*)d_in[6];
    const float* norm_w     = (const float*)d_in[7];
    const float* out_proj_w = (const float*)d_in[8];

    char* wp = (char*)d_ws;
    auto alloc = [&](size_t bytes) -> void* {
        void* p = (void*)wp;
        wp += (bytes + 255) & ~(size_t)255;
        return p;
    };
    bf16*  ubf  = (bf16*) alloc((size_t)TOKENS * DMODEL * 2);
    bf16*  wibf = (bf16*) alloc((size_t)DPROJ * DMODEL * 2);
    bf16*  wobf = (bf16*) alloc((size_t)DMODEL * DINNER * 2);
    float* zx   = (float*)alloc((size_t)TOKENS * DPROJ * 4);
    bf16*  xT   = (bf16*) alloc((size_t)NCHUNK * NHEAD * PHEAD * CHUNK * 2);
    bf16*  Bbf  = (bf16*) alloc((size_t)TOKENS * DSTATE * 2);
    bf16*  BT   = (bf16*) alloc((size_t)NCHUNK * DSTATE * CHUNK * 2);
    bf16*  Cbf  = (bf16*) alloc((size_t)TOKENS * DSTATE * 2);
    float* dtv  = (float*)alloc((size_t)TOKENS * NHEAD * 4);
    float* Acs  = (float*)alloc((size_t)NCHUNK * NHEAD * CHUNK * 4);
    float* wdec = (float*)alloc((size_t)NCHUNK * NHEAD * CHUNK * 4);
    float* cdec = (float*)alloc((size_t)NCHUNK * NHEAD * 4);
    float* dtc  = (float*)alloc((size_t)NCHUNK * NHEAD * CHUNK * 4);
    float* CB   = (float*)alloc((size_t)NCHUNK * CHUNK * CHUNK * 4);
    float* S    = (float*)alloc((size_t)NCHUNK * NHEAD * PHEAD * DSTATE * 4);
    float* P    = (float*)alloc((size_t)NCHUNK * NHEAD * PHEAD * DSTATE * 4);
    float* Y    = (float*)alloc((size_t)TOKENS * DINNER * 4);
    bf16*  ynbf = (bf16*) alloc((size_t)TOKENS * DINNER * 2);

    // 1) fp32 -> bf16 conversions
    {
        long n = (long)TOKENS * DMODEL;
        cvt_bf16_k<<<(int)((n + 255) / 256), 256, 0, stream>>>(u, ubf, n);
        n = (long)DPROJ * DMODEL;
        cvt_bf16_k<<<(int)((n + 255) / 256), 256, 0, stream>>>(in_proj_w, wibf, n);
        n = (long)DMODEL * DINNER;
        cvt_bf16_k<<<(int)((n + 255) / 256), 256, 0, stream>>>(out_proj_w, wobf, n);
    }
    // 2) in_proj GEMM: zx[8192,4384] = u @ Win^T
    gemm_bf16_k<<<dim3(TOKENS / 512, DPROJ / 32), 256, 0, stream>>>(
        ubf, wibf, zx, TOKENS, DPROJ, DMODEL);
    // 3) conv4 + SiLU + routing
    {
        long n = (long)TOKENS * CONVD;
        conv_silu_k<<<(int)((n + 255) / 256), 256, 0, stream>>>(
            zx, conv_w, conv_b, xT, Bbf, BT, Cbf);
    }
    // 4) dt softplus
    dt_k<<<(TOKENS * NHEAD) / 256, 256, 0, stream>>>(zx, dt_bias, dtv);
    // 5) cumsum of dt*A per (chunk, head)
    cumsum_k<<<NCHUNK * NHEAD, CHUNK, 0, stream>>>(dtv, A_log, Acs, wdec, cdec, dtc);
    // 6) CB = C @ B^T per chunk
    cb_k<<<dim3(NCHUNK, 32), 256, 0, stream>>>(Cbf, Bbf, CB);
    // 7) intra-chunk y = scores @ x
    y1_k<<<dim3(NCHUNK, NHEAD, 8), 256, 0, stream>>>(CB, Acs, dtc, xT, Y);
    // 8) per-chunk end states
    states_k<<<dim3(NCHUNK, NHEAD), 256, 0, stream>>>(xT, BT, wdec, S);
    // 9) inter-chunk scan
    scan_k<<<(2 * NHEAD * PHEAD * DSTATE) / 256, 256, 0, stream>>>(S, cdec, P);
    // 10) y += exp(Acs) * (C @ prevs^T)
    y2_k<<<dim3(NCHUNK, NHEAD, 8), 256, 0, stream>>>(Cbf, P, Acs, Y);
    // 11) +D*x, gate, RMSNorm -> bf16
    final_k<<<TOKENS, 256, 0, stream>>>(Y, xT, zx, Dv, norm_w, ynbf);
    // 12) out_proj GEMM -> d_out fp32
    gemm_bf16_k<<<dim3(TOKENS / 512, DMODEL / 32), 256, 0, stream>>>(
        ynbf, wobf, (float*)d_out, TOKENS, DMODEL, DINNER);
}
